// PPGN_40381282517059
// MI455X (gfx1250) — matrix operand order repaired and checked by reading, not compiled, not run on hardware
//
#include <hip/hip_runtime.h>

typedef __bf16 bf16_t;
typedef __attribute__((ext_vector_type(8)))  __bf16 v8bf;
typedef __attribute__((ext_vector_type(16))) __bf16 v16bf;
typedef __attribute__((ext_vector_type(8)))  float  v8f;

__device__ __forceinline__ bf16_t f2bf(float f) {
    unsigned u = __builtin_bit_cast(unsigned, f);
    unsigned r = (u + 0x7FFFu + ((u >> 16) & 1u)) >> 16;
    unsigned short s = (unsigned short)r;
    return __builtin_bit_cast(bf16_t, s);
}
__device__ __forceinline__ float bf2f(bf16_t b) {
    unsigned short s = __builtin_bit_cast(unsigned short, b);
    unsigned u = ((unsigned)s) << 16;
    return __builtin_bit_cast(float, u);
}

// ---------------------------------------------------------------------------
// Weight repack: fp32 [O][Kin] -> bf16 [Opad][Kpad], with optional two source
// segments (for the concat skip conv) and zero fill elsewhere (pad columns).
// ---------------------------------------------------------------------------
__global__ void pack_weight(bf16_t* __restrict__ dst, const float* __restrict__ src,
                            int O, int Opad, int Kin, int Kpad,
                            int seg1, int seg2off, int seg2len) {
    long idx = (long)blockIdx.x * blockDim.x + threadIdx.x;
    long total = (long)Opad * Kpad;
    if (idx >= total) return;
    int o = (int)(idx / Kpad), k = (int)(idx % Kpad);
    float v = 0.f;
    if (o < O) {
        if (k < seg1) v = src[(long)o * Kin + k];
        else if (k >= seg2off && k < seg2off + seg2len)
            v = src[(long)o * Kin + seg1 + (k - seg2off)];
    }
    dst[idx] = f2bf(v);
}

__global__ void zero_f32(float* p, long n) {
    long i = (long)blockIdx.x * blockDim.x + threadIdx.x;
    if (i < n) p[i] = 0.f;
}

// node_emb[n][c] = sum_f atom_emb[f][x_feats[n][f]][c]
__global__ void embed_nodes(const float* __restrict__ atom_emb,
                            const int* __restrict__ x_feats,
                            float* __restrict__ node_emb) {
    int nd = blockIdx.x;            // 0..1023
    int c  = threadIdx.x;           // 0..319
    if (c >= 300) return;
    float s = 0.f;
#pragma unroll
    for (int f = 0; f < 9; ++f) {
        int v = x_feats[nd * 9 + f];
        s += atom_emb[((long)(f * 119 + v)) * 300 + c];
    }
    node_emb[(long)nd * 300 + c] = s;
}

// scatter-add edge_data (=[1, bond_emb_sum]) into dense fp32 acc [16][4096][304]
__global__ void scatter_edges(const float* __restrict__ bond_emb,
                              const int* __restrict__ edge_index,
                              const int* __restrict__ edge_attr,
                              float* __restrict__ edge_acc) {
    int e = blockIdx.x;             // 0..8191
    int c = threadIdx.x;            // 0..319
    if (c >= 301) return;
    int src = edge_index[e], dst = edge_index[8192 + e];
    int g = src >> 6, i = src & 63, j = dst & 63;
    long row = (long)g * 4096 + j * 64 + i;   // transposed layout: s = j*64 + i
    float v;
    if (c == 0) v = 1.f;
    else {
        v = 0.f;
#pragma unroll
        for (int f = 0; f < 3; ++f) {
            int a = edge_attr[e * 3 + f];
            v += bond_emb[((long)(f * 6 + a)) * 300 + (c - 1)];
        }
    }
    atomicAdd(&edge_acc[row * 304 + c], v);
}

// Assemble z (bf16, channels-last, stride 928): ch 0..300 edges, 301..600 diag
// nodes, 601..927 zero (K padding + future mult region).
__global__ void build_z(const float* __restrict__ edge_acc,
                        const float* __restrict__ node_emb,
                        bf16_t* __restrict__ z) {
    long row = blockIdx.x;          // b*4096 + s
    int b = (int)(row >> 12);
    int s = (int)(row & 4095);
    int i = s & 63, j = s >> 6;
    for (int c = threadIdx.x; c < 928; c += 256) {
        float v = 0.f;
        if (c < 301)                 v = edge_acc[row * 304 + c];
        else if (c < 601 && i == j)  v = node_emb[(long)(b * 64 + i) * 300 + (c - 301)];
        z[row * 928 + c] = f2bf(v);
    }
}

// ---------------------------------------------------------------------------
// WMMA GEMM:  out[b][m][o] = relu( sum_k in[b][m][k] * W[o][k] + bias[o] )
// in: bf16 channels-last (stride inS, Kpad<=inS), W: packed bf16 [Opad][Kpad].
// Wave tile = 16(M) x 64(N); block = 8 waves = 128 rows of M.
// Software-pipelined: fragments for k+32 are fetched into distinct registers
// before the WMMAs for k issue, so waits drain only the older load group.
// ---------------------------------------------------------------------------
__global__ __launch_bounds__(256) void gemm_bias_act(
    const bf16_t* __restrict__ in, long inBS, int inS,
    const bf16_t* __restrict__ W, int Kpad,
    const float* __restrict__ bias,
    void* __restrict__ outp, long outBS, int outS,
    int O, int Msp, int doRelu, int outF32) {
    const int lane = threadIdx.x & 31;
    const int wave = threadIdx.x >> 5;
    const int b    = blockIdx.z;
    const int m0   = blockIdx.x * 128 + wave * 16;
    if (m0 >= Msp) return;                       // wave-uniform: EXEC stays full
    const int o0    = blockIdx.y * 64;
    const int lmod  = lane & 15;
    const int koffA = (lane < 16) ? 0 : 16;      // A frag: K half select
    const int koffB = (lane < 16) ? 0 : 16;      // B frag: K half select

    const bf16_t* inRow = in + b * inBS + (long)(m0 + lmod) * inS;
    const bf16_t* wRow  = W + (long)(o0 + lmod) * Kpad + koffB;

    auto loadA = [&](int k0) -> v16bf {
        v8bf lo = *(const v8bf*)(inRow + k0 + koffA);
        v8bf hi = *(const v8bf*)(inRow + k0 + 8 + koffA);
        return __builtin_shufflevector(lo, hi,
                 0,1,2,3,4,5,6,7,8,9,10,11,12,13,14,15);
    };
    auto loadB = [&](int jt, int k0) -> v16bf {
        return *(const v16bf*)(wRow + (long)(jt * 16) * Kpad + k0);
    };

    v8f acc[4] = {};
    v16bf aC  = loadA(0);
    v16bf bC0 = loadB(0, 0), bC1 = loadB(1, 0), bC2 = loadB(2, 0), bC3 = loadB(3, 0);
    for (int k0 = 0; k0 < Kpad; k0 += 32) {
        const int kn = (k0 + 32 < Kpad) ? k0 + 32 : 0;   // wrap keeps loop uniform
        __builtin_prefetch(inRow + k0 + 256, 0, 1);
        v16bf aN  = loadA(kn);
        v16bf bN0 = loadB(0, kn);
        v16bf bN1 = loadB(1, kn);
        v16bf bN2 = loadB(2, kn);
        v16bf bN3 = loadB(3, kn);
        acc[0] = __builtin_amdgcn_wmma_f32_16x16x32_bf16(
                     false, aC, false, bC0, (short)0, acc[0], false, false);
        acc[1] = __builtin_amdgcn_wmma_f32_16x16x32_bf16(
                     false, aC, false, bC1, (short)0, acc[1], false, false);
        acc[2] = __builtin_amdgcn_wmma_f32_16x16x32_bf16(
                     false, aC, false, bC2, (short)0, acc[2], false, false);
        acc[3] = __builtin_amdgcn_wmma_f32_16x16x32_bf16(
                     false, aC, false, bC3, (short)0, acc[3], false, false);
        aC = aN; bC0 = bN0; bC1 = bN1; bC2 = bN2; bC3 = bN3;
    }

    const int mbase = m0 + ((lane < 16) ? 0 : 8);
#pragma unroll
    for (int jt = 0; jt < 4; ++jt) {
        int o = o0 + jt * 16 + lmod;
        if (o >= O) continue;
        float bb = bias ? bias[o] : 0.f;
#pragma unroll
        for (int r = 0; r < 8; ++r) {
            int m = mbase + r;
            if (m >= Msp) continue;
            float v = acc[jt][r] + bb;
            if (doRelu && v < 0.f) v = 0.f;
            if (outF32) ((float*)outp)[b * outBS + (long)m * outS + o] = v;
            else        ((bf16_t*)outp)[b * outBS + (long)m * outS + o] = f2bf(v);
        }
    }
}

// ---------------------------------------------------------------------------
// Batched 64x64x64 matmul over spatial dims: mult[b,d] = m1[b,d] @ m2[b,d].
// One workgroup per (d, b); m1 and m2^T staged in LDS so both WMMA fragments
// are contiguous ds_load_b128 reads. Result written into the concat region.
// ---------------------------------------------------------------------------
#define LDSP 72
__global__ __launch_bounds__(256) void mult_kernel(
    const bf16_t* __restrict__ m1, const bf16_t* __restrict__ m2,
    bf16_t* __restrict__ zcat, int zS, int chOff) {
    __shared__ __attribute__((aligned(16))) bf16_t As[64][LDSP];
    __shared__ __attribute__((aligned(16))) bf16_t Bt[64][LDSP];
    const int d = blockIdx.x;
    const int b = blockIdx.y;
    const long base = (long)b * 4096 * 320 + d;
    for (int e = threadIdx.x; e < 4096; e += 256) {
        int p = e >> 6, l = e & 63;
        As[p][l] = m1[base + (long)(p * 64 + l) * 320];   // (p,l)
        Bt[l][p] = m2[base + (long)(p * 64 + l) * 320];   // (r=p,q=l) -> Bt[q][r]
    }
    __syncthreads();
    const int lane  = threadIdx.x & 31;
    const int wave  = threadIdx.x >> 5;
    const int lmod  = lane & 15;
    const int koffA = (lane < 16) ? 0 : 16;
    const int koffB = (lane < 16) ? 0 : 16;
#pragma unroll
    for (int t = 0; t < 2; ++t) {
        int tile = wave * 2 + t;
        int p0 = (tile >> 2) * 16, q0 = (tile & 3) * 16;
        v8f acc = {};
#pragma unroll
        for (int k0 = 0; k0 < 64; k0 += 32) {
            v8bf alo = *(const v8bf*)&As[p0 + lmod][k0 + koffA];
            v8bf ahi = *(const v8bf*)&As[p0 + lmod][k0 + 8 + koffA];
            v16bf af = __builtin_shufflevector(alo, ahi,
                         0,1,2,3,4,5,6,7,8,9,10,11,12,13,14,15);
            v8bf blo = *(const v8bf*)&Bt[q0 + lmod][k0 + koffB];
            v8bf bhi = *(const v8bf*)&Bt[q0 + lmod][k0 + koffB + 8];
            v16bf bfr = __builtin_shufflevector(blo, bhi,
                          0,1,2,3,4,5,6,7,8,9,10,11,12,13,14,15);
            acc = __builtin_amdgcn_wmma_f32_16x16x32_bf16(
                false, af, false, bfr, (short)0, acc, false, false);
        }
        const int mb = p0 + ((lane < 16) ? 0 : 8);
        const long ob = (long)b * 4096 * zS + chOff + d;
#pragma unroll
        for (int r = 0; r < 8; ++r) {
            int p = mb + r, q = q0 + lmod;
            zcat[ob + (long)(p * 64 + q) * zS] = f2bf(acc[r]);
        }
    }
}

// diag/offdiag mean pool of z [16][4096][640] (ch<300) -> h bf16 [16][608]
__global__ void pool_kernel(const bf16_t* __restrict__ z, bf16_t* __restrict__ h) {
    int b = blockIdx.x;
    int c = threadIdx.x;            // 0..319
    if (c < 300) {
        const bf16_t* zb = z + (long)b * 4096 * 640 + c;
        float sAll = 0.f, sDiag = 0.f;
        for (int s = 0; s < 4096; ++s) sAll += bf2f(zb[(long)s * 640]);
        for (int p = 0; p < 64; ++p)   sDiag += bf2f(zb[(long)p * 65 * 640]);
        float mDiag = sDiag * (1.f / 64.f);
        float mOff  = (sAll - sDiag) * (1.f / 4032.f);
        h[b * 608 + c]       = f2bf(mDiag);
        h[b * 608 + 300 + c] = f2bf(mOff);
    } else if (c < 308) {
        h[b * 608 + 600 + (c - 300)] = f2bf(0.f);   // K padding for fc1
    }
}

// ---------------------------------------------------------------------------
extern "C" void kernel_launch(void* const* d_in, const int* in_sizes, int n_in,
                              void* d_out, int out_size, void* d_ws, size_t ws_size,
                              hipStream_t stream) {
    (void)in_sizes; (void)n_in; (void)out_size; (void)ws_size;
    const float* atom_emb = (const float*)d_in[0];
    const float* bond_emb = (const float*)d_in[1];
    const int*   x_feats   = (const int*)d_in[46];
    const int*   edge_idx  = (const int*)d_in[47];
    const int*   edge_attr = (const int*)d_in[48];

    char* ws = (char*)d_ws;
    size_t off = 0;
    auto carve = [&](size_t bytes) { size_t o = off; off = (off + bytes + 255) & ~(size_t)255; return o; };
    bf16_t* Z0 = (bf16_t*)(ws + carve(16L * 4096 * 928 * 2));   // block1 z+cat
    bf16_t* ZA = (bf16_t*)(ws + carve(16L * 4096 * 640 * 2));   // ping
    bf16_t* ZB = (bf16_t*)(ws + carve(16L * 4096 * 640 * 2));   // pong
    bf16_t* Hb = (bf16_t*)(ws + carve(16L * 4096 * 320 * 2));   // mlp hidden
    bf16_t* M1 = (bf16_t*)(ws + carve(16L * 4096 * 320 * 2));
    bf16_t* M2 = (bf16_t*)(ws + carve(16L * 4096 * 320 * 2));
    float*  Ea = (float*)(ws + carve(16L * 4096 * 304 * 4));    // edge scatter acc
    float*  Ne = (float*)(ws + carve(1024L * 300 * 4));         // node embeddings
    bf16_t* Hp = (bf16_t*)(ws + carve(16L * 608 * 2));          // pooled features
    bf16_t* H2 = (bf16_t*)(ws + carve(16L * 320 * 2));          // fc1 out

    // ---- pack all weights to bf16 with fragment-friendly zero padding ----
    bf16_t* pw[22];
    int pi = 0;
    auto pack = [&](int srcIdx, int O, int Opad, int Kin, int Kpad,
                    int seg1, int s2off, int s2len) {
        bf16_t* dst = (bf16_t*)(ws + carve((size_t)Opad * Kpad * 2));
        long total = (long)Opad * Kpad;
        pack_weight<<<dim3((unsigned)((total + 255) / 256)), 256, 0, stream>>>(
            dst, (const float*)d_in[srcIdx], O, Opad, Kin, Kpad, seg1, s2off, s2len);
        pw[pi++] = dst;
    };
    for (int rb = 0; rb < 4; ++rb) {
        int base = 2 + rb * 10;
        int Cin = (rb == 0) ? 601 : 300;
        int Cp  = (rb == 0) ? 608 : 320;
        pack(base + 0, 300, 320, Cin,      Cp,       Cin, Cp,  0);    // mlp1 w1
        pack(base + 2, 300, 320, 300,      320,      300, 320, 0);    // mlp1 w2
        pack(base + 4, 300, 320, Cin,      Cp,       Cin, Cp,  0);    // mlp2 w1
        pack(base + 6, 300, 320, 300,      320,      300, 320, 0);    // mlp2 w2
        pack(base + 8, 300, 320, Cin + 300, Cp + 320, Cin, Cp, 300);  // skip (concat)
    }
    pack(42, 300, 320, 600, 608, 600, 608, 0);                        // fc1
    pack(44, 128, 128, 300, 320, 300, 320, 0);                        // fc2

    // ---- prologue: embeddings, scatter, z assembly ----
    embed_nodes<<<1024, 320, 0, stream>>>(atom_emb, x_feats, Ne);
    long nEa = 16L * 4096 * 304;
    zero_f32<<<dim3((unsigned)((nEa + 255) / 256)), 256, 0, stream>>>(Ea, nEa);
    scatter_edges<<<8192, 320, 0, stream>>>(bond_emb, edge_idx, edge_attr, Ea);
    build_z<<<16 * 4096, 256, 0, stream>>>(Ea, Ne, Z0);

    auto gemm = [&](const bf16_t* in, long inBS, int inS, const bf16_t* W, int Kpad,
                    const float* bias, void* out, long outBS, int outS,
                    int O, int Opad, int Msp, int batches, int relu, int f32o) {
        dim3 g((unsigned)((Msp + 127) / 128), (unsigned)(Opad / 64), (unsigned)batches);
        gemm_bias_act<<<g, 256, 0, stream>>>(in, inBS, inS, W, Kpad, bias,
                                             out, outBS, outS, O, Msp, relu, f32o);
    };

    // ---- 4 regular blocks ----
    bf16_t* bufs[5] = {Z0, ZA, ZB, ZA, ZB};   // in of rb i = bufs[i], out = bufs[i+1]
    for (int rb = 0; rb < 4; ++rb) {
        int base = 2 + rb * 10;
        int Cp = (rb == 0) ? 608 : 320;       // padded z channels
        int Sin = (rb == 0) ? 928 : 640;      // input buffer stride
        bf16_t* inZ  = bufs[rb];
        bf16_t* outZ = bufs[rb + 1];
        bf16_t** w = &pw[rb * 5];
        const float* b0 = (const float*)d_in[base + 1];
        const float* b1 = (const float*)d_in[base + 3];
        const float* b2 = (const float*)d_in[base + 5];
        const float* b3 = (const float*)d_in[base + 7];
        const float* bs = (const float*)d_in[base + 9];
        long zBS = 4096L * Sin, hBS = 4096L * 320, oBS = 4096L * 640;
        // mlp1
        gemm(inZ, zBS, Sin, w[0], Cp,  b0, Hb, hBS, 320, 300, 320, 4096, 16, 1, 0);
        gemm(Hb,  hBS, 320, w[1], 320, b1, M1, hBS, 320, 300, 320, 4096, 16, 1, 0);
        // mlp2
        gemm(inZ, zBS, Sin, w[2], Cp,  b2, Hb, hBS, 320, 300, 320, 4096, 16, 1, 0);
        gemm(Hb,  hBS, 320, w[3], 320, b3, M2, hBS, 320, 300, 320, 4096, 16, 1, 0);
        // mult -> concat region of inZ
        mult_kernel<<<dim3(300, 16), 256, 0, stream>>>(M1, M2, inZ, Sin, Cp);
        // skip conv (no relu)
        gemm(inZ, zBS, Sin, w[4], Cp + 320, bs, outZ, oBS, 640, 300, 320, 4096, 16, 0, 0);
    }

    // ---- pooling + FC head ----
    pool_kernel<<<16, 320, 0, stream>>>(ZB, Hp);
    gemm(Hp, 0, 608, pw[20], 608, (const float*)d_in[43], H2, 0, 320,
         300, 320, 16, 1, 1, 0);
    gemm(H2, 0, 320, pw[21], 320, (const float*)d_in[45], d_out, 0, 128,
         128, 128, 16, 1, 0, 1);
}
